// Attention_6966436954442
// MI455X (gfx1250) — matrix operand order repaired and checked by reading, not compiled, not run on hardware
//
#include <hip/hip_runtime.h>
#include <hip/hip_bf16.h>

#define NUM_HEADS 12
#define HD 64
#define NN 1024
#define CC 768
#define BB 16
#define EPS 1e-6f

typedef __attribute__((ext_vector_type(8)))  float        v8f;
typedef __attribute__((ext_vector_type(16))) __bf16       v16bf;
typedef __attribute__((ext_vector_type(4)))  unsigned int u32v4;
typedef __attribute__((ext_vector_type(8)))  int          i32v8;
typedef __attribute__((ext_vector_type(4)))  int          i32v4;

struct frag { u32v4 a, b; };   // 32 bytes = one 16x32 bf16 A/B fragment (8 VGPRs)

__device__ __forceinline__ unsigned short f2bf(float f) {
  unsigned u = __builtin_bit_cast(unsigned, f);
  unsigned r = u + 0x7FFFu + ((u >> 16) & 1u);   // round-to-nearest-even
  return (unsigned short)(r >> 16);
}
__device__ __forceinline__ unsigned packbf(float lo, float hi) {
  return (unsigned)f2bf(lo) | ((unsigned)f2bf(hi) << 16);
}
// A-fragment: halves [0,16) relative to p -> two b128 loads
__device__ __forceinline__ frag load_fragA(const unsigned short* p) {
  frag f; f.a = *(const u32v4*)p; f.b = *(const u32v4*)(p + 8); return f;
}
// B-fragment: halves [0,16) relative to p -> two b128 loads
__device__ __forceinline__ frag load_fragB(const unsigned short* p) {
  frag f; f.a = *(const u32v4*)p; f.b = *(const u32v4*)(p + 8); return f;
}
__device__ __forceinline__ v8f wmma_bf16(const frag& a, const frag& b, v8f c) {
  return __builtin_amdgcn_wmma_f32_16x16x32_bf16(
      false, __builtin_bit_cast(v16bf, a),
      false, __builtin_bit_cast(v16bf, b),
      (short)0, c, false, false);
}

// ---------------- DPP16 row (16-lane) reductions: row_ror:N = 0x120|N
template <int N>
__device__ __forceinline__ float dpp_ror(float x) {
  int i = __builtin_bit_cast(int, x);
  int r = __builtin_amdgcn_update_dpp(i, i, 0x120 | N, 0xF, 0xF, true);
  return __builtin_bit_cast(float, r);
}
__device__ __forceinline__ float row_max16(float x) {
  x = fmaxf(x, dpp_ror<1>(x));
  x = fmaxf(x, dpp_ror<2>(x));
  x = fmaxf(x, dpp_ror<4>(x));
  x = fmaxf(x, dpp_ror<8>(x));
  return x;
}
__device__ __forceinline__ float row_sum16(float x) {
  x += dpp_ror<1>(x);
  x += dpp_ror<2>(x);
  x += dpp_ror<4>(x);
  x += dpp_ror<8>(x);
  return x;
}

__device__ __forceinline__ unsigned lds_off(const void* p) {
  return (unsigned)(unsigned long long)p;
}

// ---------------- TDM: 2D tile (tileRows x tileCols bf16) global -> LDS
#if __has_builtin(__builtin_amdgcn_tensor_load_to_lds) && __has_builtin(__builtin_amdgcn_s_wait_tensorcnt)
#define HAVE_TDM 1
__device__ __forceinline__ void tdm_load_2d(unsigned ldsOff, const unsigned short* gsrc,
                                            int tileRows, int tileCols,
                                            int tensorCols, int tensorRows) {
  unsigned long long ga = (unsigned long long)gsrc;
  u32v4 g0;
  g0[0] = 1u;                                   // count=1 (valid), user mode
  g0[1] = ldsOff;                               // lds_addr (bytes)
  g0[2] = (unsigned)ga;                         // global_addr[31:0]
  g0[3] = (unsigned)((ga >> 32) & 0x01FFFFFFull) | 0x80000000u;  // addr[56:32] | type=2
  i32v8 g1;
  g1[0] = (int)(1u << 16);                      // data_size=1 (2 bytes/elem)
  g1[1] = (int)((unsigned)tensorCols << 16);    // tensor_dim0 [79:48] lo16
  g1[2] = (int)(((unsigned)tensorCols >> 16) & 0xFFFFu) |
          (int)(((unsigned)tensorRows & 0xFFFFu) << 16);         // dim0 hi | dim1 lo
  g1[3] = (int)(((unsigned)tensorRows >> 16) & 0xFFFFu) |
          (int)((unsigned)tileCols << 16);      // dim1 hi | tile_dim0
  g1[4] = (int)((unsigned)tileRows & 0xFFFFu);  // tile_dim1 (tile_dim2=0)
  g1[5] = (int)tensorCols;                      // tensor_dim0_stride lo32
  g1[6] = 0; g1[7] = 0;                         // stride hi, dim1_stride (unused, 2D)
  i32v4 z4 = {0, 0, 0, 0};
#if __clang_major__ >= 23
  i32v8 z8 = {0, 0, 0, 0, 0, 0, 0, 0};
  __builtin_amdgcn_tensor_load_to_lds(g0, g1, z4, z4, z8, 0);
#else
  __builtin_amdgcn_tensor_load_to_lds(g0, g1, z4, z4, 0);
#endif
}
#else
#define HAVE_TDM 0
#endif

// ---------------- Pre-pass: f32 -> bf16 (bandwidth bound, done once)
__global__ __launch_bounds__(256) void cvt_bf16_kernel(
    const float4* __restrict__ src, uint2* __restrict__ dst, int n4) {
  int i = blockIdx.x * 256 + threadIdx.x;
  if (i < n4) {
    float4 f = src[i];
    dst[i] = make_uint2(packbf(f.x, f.y), packbf(f.z, f.w));
  }
}

// ---------------- GEMM core: 256 rows (8 waves x 32) x 64 cols, B staged in LDS via TDM
template <int STORE_MODE>   // 0: route to Q/K/Vt ; 1: f32 + bias to out
__global__ __launch_bounds__(256) void gemm_kernel(
    const unsigned short* __restrict__ Abf, const unsigned short* __restrict__ Wbf,
    int ntiles, int tensorRows, const float* __restrict__ bias,
    unsigned short* __restrict__ Qb, unsigned short* __restrict__ Kb,
    unsigned short* __restrict__ Vt, float* __restrict__ out) {
  __shared__ __align__(16) unsigned short sB[64 * 384];   // 48 KB

  const int lane = threadIdx.x & 31, wid = threadIdx.x >> 5;
  const int mBlock = blockIdx.x / ntiles, nt = blockIdx.x % ntiles;
  const int mbase = (mBlock * 8 + wid) * 32;
  const int ocbase = nt * 64;
  const int lo = lane & 15, hi = lane >> 4;

  v8f acc[2][4] = {};
  const unsigned short* arow0 = Abf + (size_t)(mbase + lo) * CC + hi * 16;
  const unsigned short* arow1 = arow0 + (size_t)16 * CC;

#pragma unroll
  for (int c = 0; c < 2; ++c) {
    const unsigned short* wsrc = Wbf + (size_t)ocbase * CC + c * 384;
#if HAVE_TDM
    if (wid == 0) {
      tdm_load_2d(lds_off(sB), wsrc, 64, 384, CC, tensorRows);
      __builtin_amdgcn_s_wait_tensorcnt(0);
    }
#else
    for (int idx = threadIdx.x; idx < 64 * 384 / 8; idx += 256) {
      int row = idx / 48, col8 = (idx % 48) * 8;
      *(u32v4*)(sB + row * 384 + col8) = *(const u32v4*)(wsrc + (size_t)row * CC + col8);
    }
#endif
    __syncthreads();

    for (int k0 = c * 384; k0 < c * 384 + 384; k0 += 32) {
      frag A0 = load_fragA(arow0 + k0);
      frag A1 = load_fragA(arow1 + k0);
      const int klocal = k0 - c * 384;
#pragma unroll
      for (int s = 0; s < 4; ++s) {
        frag B = load_fragB(sB + (16 * s + lo) * 384 + klocal + hi * 16);
        acc[0][s] = wmma_bf16(A0, B, acc[0][s]);
        acc[1][s] = wmma_bf16(A1, B, acc[1][s]);
      }
    }
    __syncthreads();   // done reading sB before next chunk's DMA
  }

#pragma unroll
  for (int r = 0; r < 2; ++r) {
#pragma unroll
    for (int s = 0; s < 4; ++s) {
      int oc = ocbase + 16 * s + lo;
      float bb = (STORE_MODE == 1) ? bias[oc] : 0.f;
#pragma unroll
      for (int v = 0; v < 8; ++v) {
        int m = mbase + r * 16 + hi * 8 + v;
        float val = acc[r][s][v];
        if (STORE_MODE == 1) {
          out[(size_t)m * CC + oc] = val + bb;
        } else {
          int b = m >> 10, n = m & 1023;
          int sel = oc / CC, rr = oc % CC;
          int h = rr >> 6, d = rr & 63;
          int bh = b * NUM_HEADS + h;
          if (sel == 0)      Qb[((size_t)bh * NN + n) * HD + d] = f2bf(val * 0.125f);
          else if (sel == 1) Kb[((size_t)bh * NN + n) * HD + d] = f2bf(val);
          else               Vt[((size_t)bh * HD + d) * NN + n] = f2bf(val);
        }
      }
    }
  }
}

// ---------------- vsum[bh][d] = sum_n V[bh][n][d]
__global__ __launch_bounds__(256) void vsum_kernel(
    const unsigned short* __restrict__ Vt, float* __restrict__ vsum) {
  int i = blockIdx.x * blockDim.x + threadIdx.x;   // bh*64 + d  (exactly 12288)
  const unsigned short* p = Vt + (size_t)i * NN;
  float s = 0.f;
  for (int n = 0; n < NN; n += 2) {
    unsigned u = *(const unsigned*)(p + n);
    s += __builtin_bit_cast(float, u << 16);
    s += __builtin_bit_cast(float, u & 0xFFFF0000u);
  }
  vsum[i] = s;
}

// ---------------- fused flash attention with policy softmax
__global__ __launch_bounds__(256) void attn_kernel(
    const unsigned short* __restrict__ Qb, const unsigned short* __restrict__ Kb,
    const unsigned short* __restrict__ Vt, const float* __restrict__ policy,
    const float* __restrict__ vsum, unsigned short* __restrict__ aout) {
  __shared__ float s_policy[NN];                           // 4 KB
  __shared__ __align__(16) unsigned short sK[2][32 * 64];  // 2 x 4 KB, [key][d]
  __shared__ __align__(16) unsigned short sV[2][64 * 32];  // 2 x 4 KB, [d][key]
  __shared__ __align__(16) unsigned s_p[8][256];           // per-wave P tile (bf16)

  const int lane = threadIdx.x & 31, wid = threadIdx.x >> 5;
  const int lo = lane & 15, hi = lane >> 4;
  const int rowTile = blockIdx.x & 7;
  const int bh = blockIdx.x >> 3;
  const int b = bh / NUM_HEADS, h = bh % NUM_HEADS;

  const unsigned short* Kbh = Kb + (size_t)bh * NN * HD;
  const unsigned short* Vbh = Vt + (size_t)bh * HD * NN;

  for (int i = threadIdx.x; i < NN; i += 256) s_policy[i] = policy[(size_t)b * NN + i];

  const int rowbase = rowTile * 128 + wid * 16;
  const unsigned short* Qrow = Qb + ((size_t)bh * NN + rowbase + lo) * HD + hi * 16;
  frag Aq0 = load_fragA(Qrow);
  frag Aq1 = load_fragA(Qrow + 32);

  float m[8], ssum[8];
#pragma unroll
  for (int v = 0; v < 8; ++v) { m[v] = -3.0e38f; ssum[v] = 0.f; }
  v8f O0 = {}, O1 = {}, O2 = {}, O3 = {};
  unsigned short* pbuf = (unsigned short*)s_p[wid];

  // prologue: stage tile 0
#if HAVE_TDM
  if (wid == 0) {
    tdm_load_2d(lds_off(sK[0]), Kbh, 32, 64, 64, NN);
    tdm_load_2d(lds_off(sV[0]), Vbh, 64, 32, NN, HD);
    __builtin_amdgcn_s_wait_tensorcnt(0);
  }
#else
  for (int i = threadIdx.x; i < 256; i += 256) {
    { int r = i / 8,  c8 = (i % 8) * 8;
      *(u32v4*)(sK[0] + r * 64 + c8) = *(const u32v4*)(Kbh + (size_t)r * HD + c8); }
    { int r = i / 4,  c8 = (i % 4) * 8;
      *(u32v4*)(sV[0] + r * 32 + c8) = *(const u32v4*)(Vbh + (size_t)r * NN + c8); }
  }
#endif
  __syncthreads();

  for (int t = 0; t < NN / 32; ++t) {
    const int j0 = t * 32;
    const int cur = t & 1, nxt = cur ^ 1;
    const unsigned short* sKc = sK[cur];
    const unsigned short* sVc = sV[cur];

#if HAVE_TDM
    if (t + 1 < NN / 32 && wid == 0) {     // async-prefetch next tile via TDM
      tdm_load_2d(lds_off(sK[nxt]), Kbh + (size_t)(j0 + 32) * HD, 32, 64, 64, NN);
      tdm_load_2d(lds_off(sV[nxt]), Vbh + (j0 + 32), 64, 32, NN, HD);
    }
#endif

    v8f sc0 = {}, sc1 = {};
    {
      frag Bk;
      Bk = load_fragB(sKc + (size_t)lo * HD + hi * 16);
      sc0 = wmma_bf16(Aq0, Bk, sc0);
      Bk = load_fragB(sKc + (size_t)lo * HD + 32 + hi * 16);
      sc0 = wmma_bf16(Aq1, Bk, sc0);
      Bk = load_fragB(sKc + (size_t)(16 + lo) * HD + hi * 16);
      sc1 = wmma_bf16(Aq0, Bk, sc1);
      Bk = load_fragB(sKc + (size_t)(16 + lo) * HD + 32 + hi * 16);
      sc1 = wmma_bf16(Aq1, Bk, sc1);
    }

    const float p0 = s_policy[j0 + lo];
    const float p1 = s_policy[j0 + 16 + lo];
#pragma unroll
    for (int v = 0; v < 8; ++v) {
      float t16 = row_max16(fmaxf(sc0[v], sc1[v]));   // DPP row_ror reduction
      float mn = fmaxf(m[v], t16);
      float sc = __expf(m[v] - mn);
      m[v] = mn;
      int qrow = rowbase + hi * 8 + v;
      float e0 = __expf(sc0[v] - mn) * ((j0 + lo) == qrow ? 1.f : p0);
      float e1 = __expf(sc1[v] - mn) * ((j0 + 16 + lo) == qrow ? 1.f : p1);
      ssum[v] = ssum[v] * sc + row_sum16(e0 + e1);
      int row = hi * 8 + v;                 // C-layout -> LDS (row-major P tile)
      pbuf[row * 32 + lo]      = f2bf(e0);
      pbuf[row * 32 + 16 + lo] = f2bf(e1);
      O0[v] *= sc; O1[v] *= sc; O2[v] *= sc; O3[v] *= sc;
    }

    frag Ap = load_fragA(pbuf + lo * 32 + hi * 16);   // 2x ds_load_b128

    frag Bv;
    Bv = load_fragB(sVc + (size_t)(0 + lo) * 32 + hi * 16);
    O0 = wmma_bf16(Ap, Bv, O0);
    Bv = load_fragB(sVc + (size_t)(16 + lo) * 32 + hi * 16);
    O1 = wmma_bf16(Ap, Bv, O1);
    Bv = load_fragB(sVc + (size_t)(32 + lo) * 32 + hi * 16);
    O2 = wmma_bf16(Ap, Bv, O2);
    Bv = load_fragB(sVc + (size_t)(48 + lo) * 32 + hi * 16);
    O3 = wmma_bf16(Ap, Bv, O3);

#if HAVE_TDM
    if (wid == 0) __builtin_amdgcn_s_wait_tensorcnt(0);
    __syncthreads();                       // publish next buffer / release cur
#else
    __syncthreads();
    if (t + 1 < NN / 32) {
      const unsigned short* Kn = Kbh + (size_t)(j0 + 32) * HD;
      const unsigned short* Vn = Vbh + (j0 + 32);
      for (int i = threadIdx.x; i < 256; i += 256) {
        { int r = i / 8, c8 = (i % 8) * 8;
          *(u32v4*)(sK[nxt] + r * 64 + c8) = *(const u32v4*)(Kn + (size_t)r * HD + c8); }
        { int r = i / 4, c8 = (i % 4) * 8;
          *(u32v4*)(sV[nxt] + r * 32 + c8) = *(const u32v4*)(Vn + (size_t)r * NN + c8); }
      }
    }
    __syncthreads();
#endif
  }

  const float epsN = EPS / (float)NN;
#pragma unroll
  for (int t = 0; t < 4; ++t) {
    v8f O = (t == 0) ? O0 : (t == 1) ? O1 : (t == 2) ? O2 : O3;
    float vs = vsum[(size_t)bh * HD + t * 16 + lo];
#pragma unroll
    for (int v = 0; v < 8; ++v) {
      int n = rowbase + hi * 8 + v;
      float val = (O[v] + epsN * vs) / (ssum[v] + EPS);
      aout[((size_t)b * NN + n) * CC + h * HD + t * 16 + lo] = f2bf(val);
    }
  }
}

extern "C" void kernel_launch(void* const* d_in, const int* in_sizes, int n_in,
                              void* d_out, int out_size, void* d_ws, size_t ws_size,
                              hipStream_t stream) {
  const float* x      = (const float*)d_in[0];
  const float* policy = (const float*)d_in[1];
  const float* w_qkv  = (const float*)d_in[2];
  const float* w_proj = (const float*)d_in[3];
  const float* b_proj = (const float*)d_in[4];

  const size_t qkvElems = (size_t)BB * NUM_HEADS * NN * HD;  // 12,582,912
  unsigned short* Qb     = (unsigned short*)d_ws;
  unsigned short* Kb     = Qb + qkvElems;
  unsigned short* Vt     = Kb + qkvElems;
  unsigned short* aout   = Vt + qkvElems;
  float*          vsum   = (float*)(aout + (size_t)BB * NN * CC);
  unsigned short* xb     = (unsigned short*)(vsum + (size_t)BB * NUM_HEADS * HD);
  unsigned short* wqkvb  = xb + (size_t)BB * NN * CC;
  unsigned short* wprojb = wqkvb + (size_t)3 * CC * CC;

  const int nx4 = BB * NN * CC / 4;      // 3,145,728
  const int nq4 = 3 * CC * CC / 4;       //   442,368
  const int np4 = CC * CC / 4;           //   147,456
  cvt_bf16_kernel<<<nx4 / 256, 256, 0, stream>>>((const float4*)x, (uint2*)xb, nx4);
  cvt_bf16_kernel<<<nq4 / 256, 256, 0, stream>>>((const float4*)w_qkv, (uint2*)wqkvb, nq4);
  cvt_bf16_kernel<<<np4 / 256, 256, 0, stream>>>((const float4*)w_proj, (uint2*)wprojb, np4);

  gemm_kernel<0><<<64 * 36, 256, 0, stream>>>(xb, wqkvb, 36, 3 * CC, nullptr,
                                              Qb, Kb, Vt, nullptr);
  vsum_kernel<<<(BB * NUM_HEADS * HD) / 256, 256, 0, stream>>>(Vt, vsum);
  attn_kernel<<<BB * NUM_HEADS * 8, 256, 0, stream>>>(Qb, Kb, Vt, policy, vsum, aout);
  gemm_kernel<1><<<64 * 12, 256, 0, stream>>>(aout, wprojb, 12, CC, b_proj,
                                              nullptr, nullptr, nullptr, (float*)d_out);
}
